// MemoryBank_20933670601144
// MI455X (gfx1250) — matrix operand order repaired and checked by reading, not compiled, not run on hardware
//
#include <hip/hip_runtime.h>
#include <hip/hip_bf16.h>
#include <cmath>

// Problem constants (match reference): B=8, M=4096, D=512, S=128
#define BB 8
#define MM 4096
#define DD 512
#define SS 128

typedef __attribute__((ext_vector_type(16))) __bf16 v16bf;
typedef __attribute__((ext_vector_type(8)))  __bf16 v8bf;
typedef __attribute__((ext_vector_type(4)))  __bf16 v4bf;
typedef __attribute__((ext_vector_type(8)))  float  v8f;

#define WMMA_BF16(a, b, c) \
  __builtin_amdgcn_wmma_f32_16x16x32_bf16(false, (a), false, (b), (short)0, (c), false, false)

// ---------------------------------------------------------------------------
// WMMA operand loaders (wave32; k = 16*(lane/16) + s in both A and B)
// ---------------------------------------------------------------------------

// A 16x32 bf16 from f32 row-major source (convert on load).
__device__ __forceinline__ v16bf load_a_f32(const float* __restrict__ A, int lda,
                                            int m0, int k0, int lane) {
  const float* p = A + (size_t)(m0 + (lane & 15)) * lda + k0 + ((lane >> 4) << 4);
  v16bf r;
#pragma unroll
  for (int i = 0; i < 8; ++i) r[i] = (__bf16)p[i];
#pragma unroll
  for (int i = 0; i < 8; ++i) r[8 + i] = (__bf16)p[8 + i];
  return r;
}

// A 16x32 bf16 from packed bf16 row-major source: two contiguous 16B runs.
__device__ __forceinline__ v16bf load_a_pk(const __bf16* __restrict__ A, int lda,
                                           int m0, int k0, int lane) {
  const __bf16* p = A + (size_t)(m0 + (lane & 15)) * lda + k0 + ((lane >> 4) << 4);
  v8bf lo = *(const v8bf*)p;          // K = kb .. kb+7
  v8bf hi = *(const v8bf*)(p + 8);    // K = kb+8 .. kb+15
  v16bf r;
#pragma unroll
  for (int i = 0; i < 8; ++i) { r[i] = lo[i]; r[8 + i] = hi[i]; }
  return r;
}

// B 32x16 bf16 from f32 row-major KxN source (strided, convert on load).
__device__ __forceinline__ v16bf load_b_f32(const float* __restrict__ Bsrc, int ldb,
                                            int k0, int n0, int lane) {
  const int n  = n0 + (lane & 15);
  const int kb = k0 + ((lane >> 4) << 4);
  v16bf r;
#pragma unroll
  for (int i = 0; i < 16; ++i) r[i] = (__bf16)Bsrc[(size_t)(kb + i) * ldb + n];
  return r;
}

// B 32x16 bf16 from packed bf16 row-major KxN source (strided d16 loads).
__device__ __forceinline__ v16bf load_b_pk(const __bf16* __restrict__ Bsrc, int ldb,
                                           int k0, int n0, int lane) {
  const int n  = n0 + (lane & 15);
  const int kb = k0 + ((lane >> 4) << 4);
  v16bf r;
#pragma unroll
  for (int i = 0; i < 16; ++i) r[i] = Bsrc[(size_t)(kb + i) * ldb + n];
  return r;
}

// B 32x16 bf16 where packed source is row-major N x K (B(k,n) = Bt[n,k]):
// one contiguous 32B run per lane -> two b128 loads.
__device__ __forceinline__ v16bf load_bt_pk(const __bf16* __restrict__ Bt, int ldb,
                                            int n0, int k0, int lane) {
  const __bf16* p = Bt + (size_t)(n0 + (lane & 15)) * ldb + k0 + ((lane >> 4) << 4);
  return *(const v16bf*)p;
}

// ---------------------------------------------------------------------------
// Kernel 1a: usage_out = usage * decay + one_hot(idx) * strength
// ---------------------------------------------------------------------------
__global__ void k_usage(const float* __restrict__ usage, const float* __restrict__ decay,
                        const float* __restrict__ strength, const int* __restrict__ wi,
                        float* __restrict__ out) {
  int t = blockIdx.x * blockDim.x + threadIdx.x;
  if (t >= BB * MM) return;
  int b = t / MM, m = t % MM;
  int idx = ((wi[b] % MM) + MM) % MM;
  float u = usage[t] * decay[0];
  if (m == idx) u += strength[b];
  out[t] = u;
}

// ---------------------------------------------------------------------------
// Kernel 1b: bulk f32 -> bf16 conversion of memory into ws (float4 vectorized)
// ---------------------------------------------------------------------------
__global__ void k_membf(const float* __restrict__ mem, __bf16* __restrict__ out) {
  size_t t = (size_t)blockIdx.x * blockDim.x + threadIdx.x;  // one float4 each
  float4 v = ((const float4*)mem)[t];
  v4bf o;
  o[0] = (__bf16)v.x; o[1] = (__bf16)v.y; o[2] = (__bf16)v.z; o[3] = (__bf16)v.w;
  *(v4bf*)(out + 4 * t) = o;
}

// ---------------------------------------------------------------------------
// Kernel 1c: blended written row per batch; dual-write f32 (for exact score
// patch) and bf16 (folded into the staged memory copy -> no retrieve patch).
// ---------------------------------------------------------------------------
__global__ void k_blend(const float* __restrict__ mem, const float* __restrict__ content,
                        const float* __restrict__ strength, const float* __restrict__ temporal,
                        const int* __restrict__ wi, float* __restrict__ blend,
                        __bf16* __restrict__ membf) {
  int t = blockIdx.x * blockDim.x + threadIdx.x;
  if (t >= BB * DD) return;
  int b = t / DD, d = t % DD;
  int idx = ((wi[b] % MM) + MM) % MM;
  float a = strength[b];
  float orig = mem[((size_t)b * MM + idx) * DD + d];
  float wt   = content[b * DD + d] + temporal[(size_t)idx * DD + d];
  float val  = orig * (1.0f - a) + wt * a;
  blend[t] = val;
  membf[((size_t)b * MM + idx) * DD + d] = (__bf16)val;
}

// ---------------------------------------------------------------------------
// Kernel 2: P = (Q @ W) * 1/sqrt(D)   (einsum reassociation: 32x fewer FLOPs
// than projecting the memory bank). Dual-store f32 + bf16.
// ---------------------------------------------------------------------------
__global__ __launch_bounds__(32) void k_proj(const float* __restrict__ q,
                                             const float* __restrict__ w,
                                             float* __restrict__ p,
                                             __bf16* __restrict__ pbf) {
  const int lane = threadIdx.x;
  const int tn = blockIdx.x & 31;   // D/16 = 32 col tiles
  const int tm = blockIdx.x >> 5;   // (B*S)/16 = 64 row tiles
  v8f acc = {};
  for (int k0 = 0; k0 < DD; k0 += 32) {
    v16bf a  = load_a_f32(q, DD, tm * 16, k0, lane);
    v16bf bb = load_b_f32(w, DD, k0, tn * 16, lane);
    acc = WMMA_BF16(a, bb, acc);
  }
  const float scale = 0.044194173824159216f;  // 1/sqrt(512)
  const int n  = tn * 16 + (lane & 15);
  const int mb = tm * 16 + ((lane >> 4) << 3);
#pragma unroll
  for (int r = 0; r < 8; ++r) {
    float v = acc[r] * scale;
    p[(size_t)(mb + r) * DD + n]   = v;
    pbf[(size_t)(mb + r) * DD + n] = (__bf16)v;
  }
}

// ---------------------------------------------------------------------------
// Kernel 3: scores[b,s,m] = sum_d p[b,s,d] * membf[b,m,d]
// 16x64 tile per wave: one A operand feeds 4 WMMAs; B loads are contiguous
// 32B runs (the two lane halves read 64B of each membf row per K step).
// ---------------------------------------------------------------------------
__global__ __launch_bounds__(32) void k_scores(const __bf16* __restrict__ pbf,
                                               const __bf16* __restrict__ membf,
                                               float* __restrict__ sc) {
  const int lane = threadIdx.x;
  const int bi = blockIdx.y;
  const int tn = blockIdx.x & 63;   // M/64 = 64 col super-tiles
  const int tm = blockIdx.x >> 6;   // S/16 = 8 row tiles
  const __bf16* pb = pbf + (size_t)bi * SS * DD;
  const __bf16* mb = membf + (size_t)bi * MM * DD;
  const int n0 = tn * 64;
  v8f acc0 = {}, acc1 = {}, acc2 = {}, acc3 = {};
  for (int k0 = 0; k0 < DD; k0 += 32) {
    v16bf a  = load_a_pk(pb, DD, tm * 16, k0, lane);
    v16bf b0 = load_bt_pk(mb, DD, n0 +  0, k0, lane);
    v16bf b1 = load_bt_pk(mb, DD, n0 + 16, k0, lane);
    v16bf b2 = load_bt_pk(mb, DD, n0 + 32, k0, lane);
    v16bf b3 = load_bt_pk(mb, DD, n0 + 48, k0, lane);
    acc0 = WMMA_BF16(a, b0, acc0);
    acc1 = WMMA_BF16(a, b1, acc1);
    acc2 = WMMA_BF16(a, b2, acc2);
    acc3 = WMMA_BF16(a, b3, acc3);
  }
  float* scb = sc + (size_t)bi * SS * MM;
  const int nl   = lane & 15;
  const int mrow = tm * 16 + ((lane >> 4) << 3);
#pragma unroll
  for (int r = 0; r < 8; ++r) {
    scb[(size_t)(mrow + r) * MM + n0 +  0 + nl] = acc0[r];
    scb[(size_t)(mrow + r) * MM + n0 + 16 + nl] = acc1[r];
    scb[(size_t)(mrow + r) * MM + n0 + 32 + nl] = acc2[r];
    scb[(size_t)(mrow + r) * MM + n0 + 48 + nl] = acc3[r];
  }
}

// ---------------------------------------------------------------------------
// Kernel 3b: exact f32 patch of scores column idx[b] using the blended row
// ---------------------------------------------------------------------------
__global__ __launch_bounds__(32) void k_patch_scores(const float* __restrict__ p,
                                                     const float* __restrict__ blend,
                                                     const int* __restrict__ wi,
                                                     float* __restrict__ sc) {
  const int bi = blockIdx.x, s = blockIdx.y, lane = threadIdx.x;
  const float* pr = p + ((size_t)bi * SS + s) * DD;
  const float* bl = blend + bi * DD;
  float acc = 0.0f;
  for (int d = lane; d < DD; d += 32) acc += pr[d] * bl[d];
#pragma unroll
  for (int off = 16; off; off >>= 1) acc += __shfl_xor(acc, off, 32);
  if (lane == 0) {
    int idx = ((wi[bi] % MM) + MM) % MM;
    sc[((size_t)bi * SS + s) * MM + idx] = acc;
  }
}

// ---------------------------------------------------------------------------
// Kernel 4: exact top-k threshold (radix bisection on sortable uint keys),
// masking, softmax. One 256-thread block per (b,s) row of M=4096 scores.
// Dual-writes attn: f32 to d_out, bf16 to ws (A operand of retrieve GEMM).
// ---------------------------------------------------------------------------
__global__ __launch_bounds__(256) void k_topk_softmax(const float* __restrict__ sc,
                                                      const unsigned char* __restrict__ vmask,
                                                      const int* __restrict__ wi,
                                                      const int* __restrict__ topkp,
                                                      float* __restrict__ attn,
                                                      __bf16* __restrict__ attnbf) {
  const int bi = blockIdx.x, s = blockIdx.y, tid = threadIdx.x;
  const int K = topkp[0];
  const int idx = ((wi[bi] % MM) + MM) % MM;
  const float* row = sc + ((size_t)bi * SS + s) * MM;
  float*  arow  = attn   + ((size_t)bi * SS + s) * MM;
  __bf16* abrow = attnbf + ((size_t)bi * SS + s) * MM;

  float vals[16];
  unsigned keys[16];
  int vc = 0;
#pragma unroll
  for (int i = 0; i < 16; ++i) {
    int m = i * 256 + tid;
    bool valid = (vmask[bi * MM + m] != 0) || (m == idx);
    float v = valid ? row[m] : -__builtin_inff();
    vc += valid ? 1 : 0;
    vals[i] = v;
    unsigned u = __float_as_uint(v);
    keys[i] = (u & 0x80000000u) ? ~u : (u | 0x80000000u);  // monotone map
  }

  __shared__ int   iw[8];
  __shared__ float fw[8];

  auto reduceSumI = [&](int v) -> int {
#pragma unroll
    for (int off = 16; off; off >>= 1) v += __shfl_xor(v, off, 32);
    if ((tid & 31) == 0) iw[tid >> 5] = v;
    __syncthreads();
    int t = 0;
#pragma unroll
    for (int j = 0; j < 8; ++j) t += iw[j];
    __syncthreads();
    return t;
  };
  auto reduceMaxF = [&](float v) -> float {
#pragma unroll
    for (int off = 16; off; off >>= 1) v = fmaxf(v, __shfl_xor(v, off, 32));
    if ((tid & 31) == 0) fw[tid >> 5] = v;
    __syncthreads();
    float t = fw[0];
#pragma unroll
    for (int j = 1; j < 8; ++j) t = fmaxf(t, fw[j]);
    __syncthreads();
    return t;
  };
  auto reduceSumF = [&](float v) -> float {
#pragma unroll
    for (int off = 16; off; off >>= 1) v += __shfl_xor(v, off, 32);
    if ((tid & 31) == 0) fw[tid >> 5] = v;
    __syncthreads();
    float t = 0.0f;
#pragma unroll
    for (int j = 0; j < 8; ++j) t += fw[j];
    __syncthreads();
    return t;
  };

  // all-invalid row -> attn = 0 (uniform branch)
  if (reduceSumI(vc) == 0) {
#pragma unroll
    for (int i = 0; i < 16; ++i) {
      arow[i * 256 + tid]  = 0.0f;
      abrow[i * 256 + tid] = (__bf16)0.0f;
    }
    return;
  }

  // exact K-th largest key via 32-bit greedy bisection
  unsigned T = 0;
  for (int bit = 31; bit >= 0; --bit) {
    unsigned cand = T | (1u << bit);
    int c = 0;
#pragma unroll
    for (int i = 0; i < 16; ++i) c += (keys[i] >= cand) ? 1 : 0;
    if (reduceSumI(c) >= K) T = cand;
  }

  float mx = -__builtin_inff();
#pragma unroll
  for (int i = 0; i < 16; ++i)
    if (keys[i] >= T) mx = fmaxf(mx, vals[i]);
  mx = reduceMaxF(mx);

  float e[16];
  float se = 0.0f;
#pragma unroll
  for (int i = 0; i < 16; ++i) {
    float ev = (keys[i] >= T) ? __expf(vals[i] - mx) : 0.0f;  // kept -inf -> 0
    e[i] = ev;
    se += ev;
  }
  se = reduceSumF(se);
  float inv = 1.0f / se;
#pragma unroll
  for (int i = 0; i < 16; ++i) {
    float av = e[i] * inv;
    arow[i * 256 + tid]  = av;
    abrow[i * 256 + tid] = (__bf16)av;
  }
}

// ---------------------------------------------------------------------------
// Kernel 5: retrieved[b,s,d] = sum_m attn[b,s,m] * membf[b,m,d]
// (membf already contains the blended write-row -> no epilogue patch)
// 16x64 tile per wave, packed bf16 A, prefetch next K block of B.
// ---------------------------------------------------------------------------
__global__ __launch_bounds__(32) void k_retrieve(const __bf16* __restrict__ attnbf,
                                                 const __bf16* __restrict__ membf,
                                                 float* __restrict__ outR) {
  const int lane = threadIdx.x;
  const int bi = blockIdx.y;
  const int tn = blockIdx.x & 7;    // D/64 = 8 col super-tiles
  const int tm = blockIdx.x >> 3;   // S/16 = 8 row tiles
  const __bf16* ab = attnbf + (size_t)bi * SS * MM;
  const __bf16* mb = membf + (size_t)bi * MM * DD;
  const int n0 = tn * 64;
  v8f acc0 = {}, acc1 = {}, acc2 = {}, acc3 = {};
  for (int k0 = 0; k0 < MM; k0 += 32) {
    if (k0 + 32 < MM)  // prefetch next K block of the streamed B operand
      __builtin_prefetch(mb + (size_t)(k0 + 32 + ((lane >> 4) << 4)) * DD + n0 + (lane & 15), 0, 0);
    v16bf a  = load_a_pk(ab, MM, tm * 16, k0, lane);
    v16bf b0 = load_b_pk(mb, DD, k0, n0 +  0, lane);
    v16bf b1 = load_b_pk(mb, DD, k0, n0 + 16, lane);
    v16bf b2 = load_b_pk(mb, DD, k0, n0 + 32, lane);
    v16bf b3 = load_b_pk(mb, DD, k0, n0 + 48, lane);
    acc0 = WMMA_BF16(a, b0, acc0);
    acc1 = WMMA_BF16(a, b1, acc1);
    acc2 = WMMA_BF16(a, b2, acc2);
    acc3 = WMMA_BF16(a, b3, acc3);
  }
  float* ob = outR + (size_t)bi * SS * DD;
  const int nl = lane & 15;
  const int s0 = tm * 16 + ((lane >> 4) << 3);
#pragma unroll
  for (int r = 0; r < 8; ++r) {
    ob[(size_t)(s0 + r) * DD + n0 +  0 + nl] = acc0[r];
    ob[(size_t)(s0 + r) * DD + n0 + 16 + nl] = acc1[r];
    ob[(size_t)(s0 + r) * DD + n0 + 32 + nl] = acc2[r];
    ob[(size_t)(s0 + r) * DD + n0 + 48 + nl] = acc3[r];
  }
}

// ---------------------------------------------------------------------------
// Host-side launch
// ---------------------------------------------------------------------------
extern "C" void kernel_launch(void* const* d_in, const int* in_sizes, int n_in,
                              void* d_out, int out_size, void* d_ws, size_t ws_size,
                              hipStream_t stream) {
  (void)in_sizes; (void)n_in; (void)out_size; (void)ws_size;

  const float*         memory   = (const float*)d_in[0];
  const float*         usage    = (const float*)d_in[1];
  const float*         content  = (const float*)d_in[2];
  const float*         wstr     = (const float*)d_in[3];
  const float*         query    = (const float*)d_in[4];
  const float*         kpw      = (const float*)d_in[5];
  const float*         ttab     = (const float*)d_in[6];
  const float*         decay    = (const float*)d_in[7];
  const unsigned char* vmask    = (const unsigned char*)d_in[8];
  const int*           widx     = (const int*)d_in[9];
  const int*           topk     = (const int*)d_in[10];

  // Output layout: retrieved (B,S,D) | attn (B,S,M) | usage (B,M)
  float* outRet   = (float*)d_out;
  float* outAttn  = outRet + (size_t)BB * SS * DD;
  float* outUsage = outAttn + (size_t)BB * SS * MM;

  // Workspace layout (~62 MB):
  //   membf  bf16 (B,M,D) 33.5MB | attnbf bf16 (B,S,M) 8.4MB
  //   pbf    bf16 (B,S,D)  1MB   | blend f32 (B,D) 16KB
  //   p      f32  (B,S,D)  2MB   | sc    f32 (B,S,M) 16.8MB
  __bf16* membf  = (__bf16*)d_ws;
  __bf16* attnbf = membf + (size_t)BB * MM * DD;
  __bf16* pbf    = attnbf + (size_t)BB * SS * MM;
  float*  blend  = (float*)(pbf + (size_t)BB * SS * DD);
  float*  P      = blend + (size_t)BB * DD;
  float*  SC     = P + (size_t)BB * SS * DD;

  k_usage<<<(BB * MM + 255) / 256, 256, 0, stream>>>(usage, decay, wstr, widx, outUsage);
  k_membf<<<(BB * MM * DD / 4 + 255) / 256, 256, 0, stream>>>(memory, membf);
  k_blend<<<(BB * DD + 255) / 256, 256, 0, stream>>>(memory, content, wstr, ttab, widx,
                                                     blend, membf);
  k_proj<<<dim3((SS * BB / 16) * (DD / 16)), 32, 0, stream>>>(query, kpw, P, pbf);
  k_scores<<<dim3((SS / 16) * (MM / 64), BB), 32, 0, stream>>>(pbf, membf, SC);
  k_patch_scores<<<dim3(BB, SS), 32, 0, stream>>>(P, blend, widx, SC);
  k_topk_softmax<<<dim3(BB, SS), 256, 0, stream>>>(SC, vmask, widx, topk, outAttn, attnbf);
  k_retrieve<<<dim3((SS / 16) * (DD / 64), BB), 32, 0, stream>>>(attnbf, membf, outRet);
}
